// MPOLinearLayer_15281493639935
// MI455X (gfx1250) — matrix operand order repaired and checked by reading, not compiled, not run on hardware
//
#include <hip/hip_runtime.h>
#include <hip/hip_bf16.h>

// ---------------------------------------------------------------------------
// MPO linear layer for MI455X (gfx1250), wave32 + WMMA bf16.
//
//   y[8192,4096] = x @ W^T + bias,  W[(m1,m2),(n1,n2)] = sum_r c1[m1,n1,r]*c2[r,m2,n2]
//
// Strategy: materialize T = c1_flat[4096x256] @ c2^T[4096x256]^T (8.6 GFLOP),
// then y = x_bf16 @ T (index-permuted as W) (275 GFLOP) with
// v_wmma_f32_16x16x32_bf16.  The (m1,n1,m2,n2)->(m1,m2,n1,n2) permute is
// folded into the B-operand address math of the main GEMM.
// ---------------------------------------------------------------------------

typedef __attribute__((ext_vector_type(16))) __bf16 v16bf;
typedef __attribute__((ext_vector_type(8)))  float  v8f;

union BF16Frag {
    v16bf v;
    uint4 q[2];
};

__device__ __forceinline__ unsigned short f2bf(float f) {
    // round-to-nearest-even f32 -> bf16
    unsigned u = __builtin_bit_cast(unsigned, f);
    unsigned r = u + 0x7FFFu + ((u >> 16) & 1u);
    return (unsigned short)(r >> 16);
}

// --------------------------- conversion kernels ----------------------------

// f32 -> bf16, 4 elements per thread (float4 in, uint2 out)
__global__ __launch_bounds__(256) void cvt_f32_bf16_kernel(
    const float* __restrict__ in, unsigned short* __restrict__ out, int n4)
{
    int i = blockIdx.x * 256 + threadIdx.x;
    if (i >= n4) return;
    float4 f = ((const float4*)in)[i];
    uint2 p;
    p.x = (unsigned)f2bf(f.x) | ((unsigned)f2bf(f.y) << 16);
    p.y = (unsigned)f2bf(f.z) | ((unsigned)f2bf(f.w) << 16);
    ((uint2*)out)[i] = p;
}

// core2 f32 [256][4096] -> bf16 [4096][256]  (row = (m2,n2), col = r)
// LDS-tiled 32x32 transpose, block (32,8), grid (4096/32, 256/32)
__global__ __launch_bounds__(256) void transpose_cvt_core2_kernel(
    const float* __restrict__ in, unsigned short* __restrict__ out)
{
    __shared__ float tile[32][33];
    const int tx = threadIdx.x;      // 0..31
    const int ty = threadIdx.y;      // 0..7
    const int j0 = blockIdx.x * 32;  // (m2,n2) tile base
    const int r0 = blockIdx.y * 32;  // r tile base
#pragma unroll
    for (int i = 0; i < 4; ++i)
        tile[ty + i * 8][tx] = in[(size_t)(r0 + ty + i * 8) * 4096 + j0 + tx];
    __syncthreads();
#pragma unroll
    for (int i = 0; i < 4; ++i)
        out[(size_t)(j0 + ty + i * 8) * 256 + r0 + tx] = f2bf(tile[tx][ty + i * 8]);
}

// ------------------------- WMMA fragment loaders ---------------------------
// A-frag (16x32, M x K), row-major source with leading dim ldA (elements):
//   lane l (l15 = l&15, lh = l>>4) holds row m0+l15,
//   chunk0 = K[lh*16 .. lh*16+7], chunk1 = K[lh*16+8 .. lh*16+15]   (2 x b128)
__device__ __forceinline__ v16bf load_a_frag(const unsigned short* __restrict__ A,
                                             int ldA, int row, int kb, int lh)
{
    const unsigned short* p = A + (size_t)row * ldA + kb + lh * 16;
    BF16Frag f;
    f.q[0] = *(const uint4*)(p);
    f.q[1] = *(const uint4*)(p + 8);
    return f.v;
}

// B-frag (32x16, K x N) where B[k][n] = W[n][k] (row-major W, ld elements):
//   lane l holds column n0+l15, K[lh*16 .. lh*16+15]  (contiguous 32B)
__device__ __forceinline__ v16bf load_b_frag_rowmajor(const unsigned short* __restrict__ W,
                                                      int ldW, int row, int kb, int lh)
{
    const unsigned short* p = W + (size_t)row * ldW + kb + lh * 16;
    BF16Frag f;
    f.q[0] = *(const uint4*)(p);
    f.q[1] = *(const uint4*)(p + 8);
    return f.v;
}

// ------------------------- stage 1: T = c1 @ c2^T --------------------------
// A = core1 bf16 [4096][256], B = core2^T bf16 [4096][256], T bf16 [4096][4096]
// grid (4096/64, 4096/256), block 256 (8 waves, each wave: 32x64 tile)
__global__ __launch_bounds__(256) void mpo_weight_gemm_kernel(
    const unsigned short* __restrict__ A,
    const unsigned short* __restrict__ B,
    unsigned short* __restrict__ T)
{
    const int lane = threadIdx.x & 31;
    const int wave = threadIdx.x >> 5;
    const int l15  = lane & 15;
    const int lh   = lane >> 4;
    const int m0 = blockIdx.x * 64  + (wave >> 2) * 32;
    const int n0 = blockIdx.y * 256 + (wave & 3) * 64;

    v8f acc[2][4] = {};

    for (int kb = 0; kb < 256; kb += 32) {
        v16bf a[2], b[4];
#pragma unroll
        for (int i = 0; i < 2; ++i)
            a[i] = load_a_frag(A, 256, m0 + i * 16 + l15, kb, lh);
#pragma unroll
        for (int j = 0; j < 4; ++j)
            b[j] = load_b_frag_rowmajor(B, 256, n0 + j * 16 + l15, kb, lh);
#pragma unroll
        for (int i = 0; i < 2; ++i)
#pragma unroll
            for (int j = 0; j < 4; ++j)
                acc[i][j] = __builtin_amdgcn_wmma_f32_16x16x32_bf16(
                    false, a[i], false, b[j], (short)0, acc[i][j], false, false);
    }

#pragma unroll
    for (int i = 0; i < 2; ++i)
#pragma unroll
        for (int j = 0; j < 4; ++j) {
            const int col = n0 + j * 16 + l15;
#pragma unroll
            for (int v = 0; v < 8; ++v)
                T[(size_t)(m0 + i * 16 + v + 8 * lh) * 4096 + col] = f2bf(acc[i][j][v]);
        }
}

// ---------------- stage 2: y = x_bf16 @ T(perm) + bias ---------------------
// X bf16 [8192][4096]; T bf16 [4096][4096] accessed as W via
//   addr(o,k) = ((o>>6)*64 + (k>>6))*4096 + (o&63)*64 + (k&63)
// grid (8192/64, 4096/256), block 256 (8 waves, each wave: 32x64 tile)
__global__ __launch_bounds__(256) void mpo_main_gemm_kernel(
    const unsigned short* __restrict__ X,
    const unsigned short* __restrict__ T,
    const float* __restrict__ bias,
    float* __restrict__ Y)
{
    const int lane = threadIdx.x & 31;
    const int wave = threadIdx.x >> 5;
    const int l15  = lane & 15;
    const int lh   = lane >> 4;
    const int m0 = blockIdx.x * 64  + (wave >> 2) * 32;
    const int n0 = blockIdx.y * 256 + (wave & 3) * 64;

    // per-lane permuted base offset for each B row o = n0 + j*16 + l15
    int brow[4];
#pragma unroll
    for (int j = 0; j < 4; ++j) {
        const int o = n0 + j * 16 + l15;
        brow[j] = (o >> 6) * (64 * 4096) + (o & 63) * 64;
    }

    v8f acc[2][4] = {};

    for (int kb = 0; kb < 4096; kb += 32) {
        v16bf a[2], b[4];
#pragma unroll
        for (int i = 0; i < 2; ++i)
            a[i] = load_a_frag(X, 4096, m0 + i * 16 + l15, kb, lh);

        const int k = kb + lh * 16;          // 16-aligned, stays inside 64-block
        const int koff = (k >> 6) * 4096 + (k & 63);
#pragma unroll
        for (int j = 0; j < 4; ++j) {
            const unsigned short* p = T + brow[j] + koff;
            BF16Frag f;
            f.q[0] = *(const uint4*)(p);
            f.q[1] = *(const uint4*)(p + 8);
            b[j] = f.v;
        }
#pragma unroll
        for (int i = 0; i < 2; ++i)
#pragma unroll
            for (int j = 0; j < 4; ++j)
                acc[i][j] = __builtin_amdgcn_wmma_f32_16x16x32_bf16(
                    false, a[i], false, b[j], (short)0, acc[i][j], false, false);
    }

#pragma unroll
    for (int j = 0; j < 4; ++j) {
        const int col = n0 + j * 16 + l15;
        const float bv = bias[col];
#pragma unroll
        for (int i = 0; i < 2; ++i)
#pragma unroll
            for (int v = 0; v < 8; ++v)
                Y[(size_t)(m0 + i * 16 + v + 8 * lh) * 4096 + col] = acc[i][j][v] + bv;
    }
}

// ------------------------------- launcher ----------------------------------

extern "C" void kernel_launch(void* const* d_in, const int* in_sizes, int n_in,
                              void* d_out, int out_size, void* d_ws, size_t ws_size,
                              hipStream_t stream) {
    const float* x     = (const float*)d_in[0];  // [8192*4096]
    const float* core1 = (const float*)d_in[1];  // [64*64*256]  = [4096][256]
    const float* core2 = (const float*)d_in[2];  // [256*64*64]  = [256][4096]
    const float* bias  = (const float*)d_in[3];  // [4096]
    float* y = (float*)d_out;                    // [8192*4096]

    char* ws = (char*)d_ws;
    unsigned short* xb  = (unsigned short*)(ws);                              // 64 MB
    unsigned short* a1  = (unsigned short*)(ws + (size_t)64 * 1024 * 1024);   //  2 MB
    unsigned short* c2t = (unsigned short*)(ws + (size_t)66 * 1024 * 1024);   //  2 MB
    unsigned short* tw  = (unsigned short*)(ws + (size_t)68 * 1024 * 1024);   // 32 MB

    // 1) x -> bf16 (33554432 elems = 8388608 float4s)
    cvt_f32_bf16_kernel<<<8388608 / 256, 256, 0, stream>>>(x, xb, 8388608);
    // 2) core1 -> bf16 (1048576 elems = 262144 float4s)
    cvt_f32_bf16_kernel<<<262144 / 256, 256, 0, stream>>>(core1, a1, 262144);
    // 3) core2 [256][4096] -> transpose+convert -> [4096][256] bf16
    transpose_cvt_core2_kernel<<<dim3(128, 8), dim3(32, 8), 0, stream>>>(core2, c2t);
    // 4) T = core1 @ core2^T   (4096 x 4096 x 256, bf16 WMMA)
    mpo_weight_gemm_kernel<<<dim3(64, 16), 256, 0, stream>>>(a1, c2t, tw);
    // 5) y = x @ W^T + bias    (8192 x 4096 x 4096, bf16 WMMA, permuted B index)
    mpo_main_gemm_kernel<<<dim3(128, 16), 256, 0, stream>>>(xb, tw, bias, y);
}
